// ReformerDecoderLayer_19164144075421
// MI455X (gfx1250) — matrix operand order repaired and checked by reading, not compiled, not run on hardware
//
#include <hip/hip_runtime.h>
#include <hip/hip_bf16.h>
#include <math.h>

// ---------------------------------------------------------------------------
// Reformer decoder layer for MI455X (gfx1250).
// All GEMMs via v_wmma_f32_16x16x32_f16, f16 operands staged in LDS so every
// fragment load is a ds_load_b128. GEMM waves own 32x32 register tiles
// (2x2 WMMA accumulators) for 4 WMMAs per K-step with fragment reuse.
// B=4, S=4096, D=512, H=8, DK=DV=64, BUCKET=64, NB=64
// ---------------------------------------------------------------------------

#define RB 4
#define RS 4096
#define RD 512
#define RH 8
#define RDK 64
#define RBH (RB * RH)       // 32
#define RBS (RB * RS)       // 16384
#define RNC 64              // chunks per (b,h)

typedef __attribute__((ext_vector_type(16))) _Float16 v16h;
typedef __attribute__((ext_vector_type(8)))  _Float16 v8h;
typedef __attribute__((ext_vector_type(4)))  _Float16 v4h;
typedef __attribute__((ext_vector_type(8)))  float    v8f;

// ---------------- WMMA fragment helpers (layouts per CDNA5 ISA) --

__device__ __forceinline__ v8f wmma_f16(v16h a, v16h b, v8f c) {
  return __builtin_amdgcn_wmma_f32_16x16x32_f16(
      /*neg_a=*/false, a, /*neg_b=*/false, b,
      /*c_mod=*/(short)0, c, /*reuse_a=*/false, /*reuse_b=*/false);
}

__device__ __forceinline__ v16h frag_cat(v8h lo, v8h hi) {
  return __builtin_shufflevector(lo, hi, 0, 1, 2, 3, 4, 5, 6, 7,
                                 8, 9, 10, 11, 12, 13, 14, 15);
}

// A matrix 16x32 (MxK), row-major in LDS, stride ld (halves; ld*2 % 16 == 0).
// lanes 0-15: row=lane, K {0..7,16..23}; lanes 16-31: row=lane-16, K {8..15,24..31}
// -> two contiguous 8-half (16B) runs per lane.
__device__ __forceinline__ v16h frag_a(const _Float16* p, int ld, int row0, int k0, int lane) {
  int m  = row0 + (lane & 15);
  int kb = (lane >> 4) << 3;                   // 0 or 8
  const _Float16* rp = p + m * ld + k0 + kb;
  v8h lo = *(const v8h*)rp;                    // K = kb .. kb+7
  v8h hi = *(const v8h*)(rp + 16);             // K = kb+16 .. kb+23
  return frag_cat(lo, hi);
}

// B matrix 32x16 (KxN) sourced from an NxK row-major buffer (stride ld halves):
// lane supplies column n = n0+(lane&15), K = k0 + 8*(lane>>4) + {0..7,16..23}
// -> two contiguous 8-half (16B) runs per lane.
__device__ __forceinline__ v16h frag_bt(const _Float16* p, int ld, int n0, int k0, int lane) {
  int n  = n0 + (lane & 15);
  int kb = k0 + ((lane >> 4) << 3);            // +0 or +8
  const _Float16* rp = p + n * ld + kb;
  v8h lo = *(const v8h*)rp;
  v8h hi = *(const v8h*)(rp + 16);
  return frag_cat(lo, hi);
}

// ---------------- f32 -> f16 conversion (element-wise, 4/thread) -----------
__global__ __launch_bounds__(256) void cvt_kernel(
    const float* __restrict__ src, _Float16* __restrict__ dst, int n4) {
  int i = blockIdx.x * 256 + threadIdx.x;
  if (i >= n4) return;
  float4 v = ((const float4*)src)[i];
  v4h h = {(_Float16)v.x, (_Float16)v.y, (_Float16)v.z, (_Float16)v.w};
  ((v4h*)dst)[i] = h;
}

// ---------------- Generic WMMA GEMM ----------------------------------------
// A16: [M,K] f16, B16: [K,N] f16. Optional f32 C (with bias/resid/relu) and
// f16 C16 outputs. 256 threads = 8 waves (2x4 wave grid); workgroup tile
// 64(M) x 128(N); each wave: 32x32 register tile = 2x2 WMMA accs; K-step 32.
__global__ __launch_bounds__(256) void gemm_wmma_kernel(
    const _Float16* __restrict__ A, const _Float16* __restrict__ Bw,
    const float* __restrict__ bias, const float* __restrict__ resid,
    float* __restrict__ C, _Float16* __restrict__ C16,
    int M, int N, int K, int relu) {
  __shared__ _Float16 As[64][40];     // [m][k]
  __shared__ _Float16 BsT[128][40];   // [n][k]  (transposed -> contiguous frags)
  int tid  = threadIdx.x;
  int lane = tid & 31;
  int wave = tid >> 5;
  int tileM = blockIdx.y * 64;
  int tileN = blockIdx.x * 128;
  int wm0 = (wave >> 2) * 32;          // 0 or 32
  int wn0 = (wave & 3) * 32;           // 0..96

  v8f acc[2][2];
#pragma unroll
  for (int mi = 0; mi < 2; ++mi)
#pragma unroll
    for (int ni = 0; ni < 2; ++ni)
      acc[mi][ni] = (v8f){0.f,0.f,0.f,0.f,0.f,0.f,0.f,0.f};

  int am  = tid >> 2, ak = (tid & 3) * 8;      // A: 8 halves per thread
  int bk2 = tid >> 3, bn = (tid & 7) * 16;     // B: 16 halves per thread

  for (int k0 = 0; k0 < K; k0 += 32) {
    *(v8h*)&As[am][ak] = *(const v8h*)&A[(size_t)(tileM + am) * K + k0 + ak];
    v8h b0 = *(const v8h*)&Bw[(size_t)(k0 + bk2) * N + tileN + bn];
    v8h b1 = *(const v8h*)&Bw[(size_t)(k0 + bk2) * N + tileN + bn + 8];
#pragma unroll
    for (int i = 0; i < 8; ++i) {
      BsT[bn + i][bk2]     = b0[i];
      BsT[bn + 8 + i][bk2] = b1[i];
    }
    if (k0 + 32 < K) {
      __builtin_prefetch(&A[(size_t)(tileM + am) * K + k0 + 32 + ak], 0, 1);
      __builtin_prefetch(&Bw[(size_t)(k0 + 32 + bk2) * N + tileN + bn], 0, 1);
    }
    __syncthreads();
    v16h a0 = frag_a(&As[0][0], 40, wm0, 0, lane);
    v16h a1 = frag_a(&As[0][0], 40, wm0 + 16, 0, lane);
    v16h bb0 = frag_bt(&BsT[0][0], 40, wn0, 0, lane);
    v16h bb1 = frag_bt(&BsT[0][0], 40, wn0 + 16, 0, lane);
    acc[0][0] = wmma_f16(a0, bb0, acc[0][0]);
    acc[0][1] = wmma_f16(a0, bb1, acc[0][1]);
    acc[1][0] = wmma_f16(a1, bb0, acc[1][0]);
    acc[1][1] = wmma_f16(a1, bb1, acc[1][1]);
    __syncthreads();
  }

  int hl = lane >> 4, cl = lane & 15;
#pragma unroll
  for (int mi = 0; mi < 2; ++mi) {
    int rbase = tileM + wm0 + mi * 16 + (hl << 3);
#pragma unroll
    for (int ni = 0; ni < 2; ++ni) {
      int col = tileN + wn0 + ni * 16 + cl;
      float bv = bias ? bias[col] : 0.f;
#pragma unroll
      for (int r = 0; r < 8; ++r) {
        int row = rbase + r;
        float v = acc[mi][ni][r] + bv;
        if (resid) v += resid[(size_t)row * N + col];
        if (relu)  v = fmaxf(v, 0.f);
        if (C)   C[(size_t)row * N + col]   = v;
        if (C16) C16[(size_t)row * N + col] = (_Float16)v;
      }
    }
  }
}

// ---------------- LSH hashing: bucket = argmax([r,-r]), 64x32 rotation -----
__global__ __launch_bounds__(256) void hash_kernel(
    const _Float16* __restrict__ T16, const float* __restrict__ rot,
    int* __restrict__ out) {
  __shared__ float rl[64 * 32];
  int bh = blockIdx.x;
  int b = bh >> 3, h = bh & 7;
  for (int i = threadIdx.x; i < 64 * 32; i += 256) rl[i] = rot[h * 2048 + i];
  __syncthreads();
  int s = blockIdx.y * 256 + threadIdx.x;
  const _Float16* q = T16 + (size_t)(b * RS + s) * RD + h * RDK;
  float qv[64];
#pragma unroll
  for (int kb = 0; kb < 64; kb += 8) {
    v8h t = *(const v8h*)(q + kb);
#pragma unroll
    for (int i = 0; i < 8; ++i) qv[kb + i] = (float)t[i];
  }
  float best = -3.0e38f;
  int bi = 0;
  for (int j = 0; j < 32; ++j) {
    float r = 0.f;
#pragma unroll
    for (int k = 0; k < 64; ++k) r += qv[k] * rl[k * 32 + j];
    if (r > best)  { best = r;  bi = j; }
    if (-r > best) { best = -r; bi = j + 32; }
  }
  out[(size_t)bh * RS + s] = bi;
}

// ---------------- Stable counting sort by bucket (== argsort(b*S+pos)) -----
__global__ __launch_bounds__(64) void sort_kernel(
    const int* __restrict__ buckets, int* __restrict__ perm) {
  __shared__ int cnt[64][65];
  __shared__ int base[64];
  int bh = blockIdx.x, t = threadIdx.x;
  const int* bb = buckets + (size_t)bh * RS;
  for (int k = 0; k < 64; ++k) cnt[t][k] = 0;
  __syncthreads();
  for (int i = 0; i < 64; ++i) cnt[t][bb[t * 64 + i]]++;
  __syncthreads();
  {
    int run = 0;
    for (int tt = 0; tt < 64; ++tt) { int c = cnt[tt][t]; cnt[tt][t] = run; run += c; }
    base[t] = run;
  }
  __syncthreads();
  if (t == 0) { int run = 0; for (int k = 0; k < 64; ++k) { int c = base[k]; base[k] = run; run += c; } }
  __syncthreads();
  for (int i = 0; i < 64; ++i) {
    int s = t * 64 + i;
    int bk = bb[s];
    int rank = base[bk] + cnt[t][bk];
    cnt[t][bk]++;
    perm[(size_t)bh * RS + rank] = s;
  }
}

// ---------------- Chunked LSH attention ------------------------------------
// One workgroup (128 threads = 4 waves) per (b,h,chunk). Gathers q/k/v rows
// straight through the permutation (fused gather), computes
// scores = cq(64x64).ckx(128x64)^T -> mask -> register softmax -> O = attn.cvx,
// scatters O through perm_q (fused unsort) as f16. A-fragments are reused
// across all column tiles (ks-outer loop order).
__global__ __launch_bounds__(128) void attn_kernel(
    const _Float16* __restrict__ Q16, const _Float16* __restrict__ K16,
    const _Float16* __restrict__ V16, const int* __restrict__ bq,
    const int* __restrict__ bk, const int* __restrict__ pq,
    const int* __restrict__ pk, _Float16* __restrict__ O16) {
  __shared__ _Float16 cq_l[64][72];     // [row][k]
  __shared__ _Float16 kx_l[128][72];    // [j][k]   (frag_bt-ready)
  __shared__ _Float16 vxT[64][136];     // [dv][j]  (frag_bt-ready)
  __shared__ _Float16 at_l[64][136];    // [row][j] (frag_a-ready)
  __shared__ int pq_l[64];
  __shared__ int pk_l[128];
  __shared__ int bq_l[64];
  __shared__ int bkx_l[128];

  int tid = threadIdx.x, lane = tid & 31, wave = tid >> 5;
  int blk = blockIdx.x;
  int bh = blk >> 6, c = blk & 63;
  int b = bh >> 3, h = bh & 7;
  size_t bhS = (size_t)bh * RS;
  int cprev = (c + RNC - 1) & (RNC - 1);

  if (tid < 64) pq_l[tid] = pq[bhS + c * 64 + tid];
  {
    int j = tid;   // exactly 128 threads
    int srt = (j < 64) ? (cprev * 64 + j) : (c * 64 + (j - 64));
    pk_l[j] = pk[bhS + srt];
  }
  __syncthreads();
  if (tid < 64) bq_l[tid] = bq[bhS + pq_l[tid]];
  bkx_l[tid] = bk[bhS + pk_l[tid]];

  for (int i = tid; i < 64 * 16; i += 128) {
    int r = i >> 4, k4 = (i & 15) * 4;
    *(v4h*)&cq_l[r][k4] =
        *(const v4h*)(Q16 + (size_t)(b * RS + pq_l[r]) * RD + h * RDK + k4);
  }
  for (int i = tid; i < 128 * 16; i += 128) {
    int j = i >> 4, k4 = (i & 15) * 4;
    size_t rowoff = (size_t)(b * RS + pk_l[j]) * RD + h * RDK + k4;
    *(v4h*)&kx_l[j][k4] = *(const v4h*)(K16 + rowoff);
    v4h vt = *(const v4h*)(V16 + rowoff);
#pragma unroll
    for (int d = 0; d < 4; ++d) vxT[k4 + d][j] = vt[d];
  }
  __syncthreads();

  // ---- phase 1: scores; A-fragment loaded once per ks, reused over 8 tiles
  int wm = wave * 16;
  v8f sacc[8];
#pragma unroll
  for (int t = 0; t < 8; ++t) sacc[t] = (v8f){0.f,0.f,0.f,0.f,0.f,0.f,0.f,0.f};
#pragma unroll
  for (int ks = 0; ks < 64; ks += 32) {
    v16h af = frag_a(&cq_l[0][0], 72, wm, ks, lane);
#pragma unroll
    for (int t = 0; t < 8; ++t) {
      v16h bf = frag_bt(&kx_l[0][0], 72, t * 16, ks, lane);
      sacc[t] = wmma_f16(af, bf, sacc[t]);
    }
  }

  // ---- mask + softmax in registers (row's 128 cols live in one 16-lane half)
  int cl = lane & 15, hl = lane >> 4;
#pragma unroll
  for (int r = 0; r < 8; ++r) {
    int row = wm + r + 8 * hl;
    int bqr = bq_l[row];
    float mx = -3.0e38f;
#pragma unroll
    for (int t = 0; t < 8; ++t) {
      float v = sacc[t][r] * 0.125f;                 // dk^-0.5
      if (bqr != bkx_l[t * 16 + cl]) v = -1.0e9f;
      sacc[t][r] = v;
      mx = fmaxf(mx, v);
    }
#pragma unroll
    for (int off = 1; off < 16; off <<= 1) mx = fmaxf(mx, __shfl_xor(mx, off, 32));
    float sum = 0.f;
#pragma unroll
    for (int t = 0; t < 8; ++t) {
      float e = __expf(sacc[t][r] - mx);
      sacc[t][r] = e;
      sum += e;
    }
#pragma unroll
    for (int off = 1; off < 16; off <<= 1) sum += __shfl_xor(sum, off, 32);
    float inv = 1.0f / sum;
#pragma unroll
    for (int t = 0; t < 8; ++t) at_l[row][t * 16 + cl] = (_Float16)(sacc[t][r] * inv);
  }
  __syncthreads();

  // ---- phase 2: O = attn(16x128) x cvx(128x64); A-fragment reused over 4 nt
  v8f oacc[4];
#pragma unroll
  for (int nt = 0; nt < 4; ++nt) oacc[nt] = (v8f){0.f,0.f,0.f,0.f,0.f,0.f,0.f,0.f};
#pragma unroll
  for (int ks = 0; ks < 128; ks += 32) {
    v16h af = frag_a(&at_l[0][0], 136, wm, ks, lane);
#pragma unroll
    for (int nt = 0; nt < 4; ++nt) {
      v16h bf = frag_bt(&vxT[0][0], 136, nt * 16, ks, lane);
      oacc[nt] = wmma_f16(af, bf, oacc[nt]);
    }
  }
  int rb = wm + (hl << 3);
#pragma unroll
  for (int nt = 0; nt < 4; ++nt) {
    int col = nt * 16 + cl;
#pragma unroll
    for (int r = 0; r < 8; ++r) {
      int sorig = pq_l[rb + r];
      O16[(size_t)(b * RS + sorig) * RD + h * RDK + col] = (_Float16)oacc[nt][r];
    }
  }
}

// ---------------- LayerNorm (+ optional f16 copy for next GEMM) ------------
__global__ __launch_bounds__(256) void ln_kernel(
    const float* __restrict__ X, const float* __restrict__ g,
    const float* __restrict__ bt, float* __restrict__ Y,
    _Float16* __restrict__ Y16) {
  __shared__ float red[256];
  __shared__ float stat;
  int row = blockIdx.x, tid = threadIdx.x;
  const float* x = X + (size_t)row * RD;
  float v0 = x[tid], v1 = x[tid + 256];
  red[tid] = v0 + v1;
  __syncthreads();
  for (int o = 128; o > 0; o >>= 1) { if (tid < o) red[tid] += red[tid + o]; __syncthreads(); }
  if (tid == 0) stat = red[0] * (1.0f / RD);
  __syncthreads();
  float mean = stat;
  float d0 = v0 - mean, d1 = v1 - mean;
  red[tid] = d0 * d0 + d1 * d1;
  __syncthreads();
  for (int o = 128; o > 0; o >>= 1) { if (tid < o) red[tid] += red[tid + o]; __syncthreads(); }
  if (tid == 0) stat = red[0] * (1.0f / RD);
  __syncthreads();
  float inv = rsqrtf(stat + 1e-5f);
  float y0 = d0 * inv * g[tid]       + bt[tid];
  float y1 = d1 * inv * g[tid + 256] + bt[tid + 256];
  Y[(size_t)row * RD + tid]       = y0;
  Y[(size_t)row * RD + tid + 256] = y1;
  if (Y16) {
    Y16[(size_t)row * RD + tid]       = (_Float16)y0;
    Y16[(size_t)row * RD + tid + 256] = (_Float16)y1;
  }
}

// ---------------------------------------------------------------------------
// Host orchestration
// ---------------------------------------------------------------------------

static void launch_gemm(const _Float16* A, const _Float16* Bw, const float* bias,
                        const float* resid, float* C, _Float16* C16,
                        int M, int N, int K, int relu, hipStream_t s) {
  dim3 grid(N / 128, M / 64), blk(256);
  gemm_wmma_kernel<<<grid, blk, 0, s>>>(A, Bw, bias, resid, C, C16, M, N, K, relu);
}

struct Ws {
  _Float16 *dec16, *enc16;
  _Float16 *Q16, *K16, *V16, *O16, *Hb16;
  float *X1, *X2, *TMP;
  _Float16 *X1_16, *X2_16;
  int *bq, *bk, *pq, *pk;
};

static void run_lsh_attention(const _Float16* xq16, const _Float16* xkv16,
                              const float* xq_resid,
                              const _Float16* Wq16, const _Float16* Wk16,
                              const _Float16* Wv16, const _Float16* Wo16,
                              const float* g, const float* bvec, const float* rot,
                              float* xout, _Float16* xout16, const Ws& w,
                              hipStream_t s) {
  launch_gemm(xq16,  Wq16, nullptr, nullptr, nullptr, w.Q16, RBS, RD, RD, 0, s);
  launch_gemm(xkv16, Wk16, nullptr, nullptr, nullptr, w.K16, RBS, RD, RD, 0, s);
  launch_gemm(xkv16, Wv16, nullptr, nullptr, nullptr, w.V16, RBS, RD, RD, 0, s);
  hash_kernel<<<dim3(RBH, RS / 256), 256, 0, s>>>(w.Q16, rot, w.bq);
  hash_kernel<<<dim3(RBH, RS / 256), 256, 0, s>>>(w.K16, rot, w.bk);
  sort_kernel<<<RBH, 64, 0, s>>>(w.bq, w.pq);
  sort_kernel<<<RBH, 64, 0, s>>>(w.bk, w.pk);
  attn_kernel<<<RBH * RNC, 128, 0, s>>>(w.Q16, w.K16, w.V16, w.bq, w.bk,
                                        w.pq, w.pk, w.O16);
  launch_gemm(w.O16, Wo16, nullptr, /*resid=*/xq_resid, w.TMP, nullptr,
              RBS, RD, RD, 0, s);
  ln_kernel<<<RBS, 256, 0, s>>>(w.TMP, g, bvec, xout, xout16);
}

extern "C" void kernel_launch(void* const* d_in, const int* in_sizes, int n_in,
                              void* d_out, int out_size, void* d_ws, size_t ws_size,
                              hipStream_t stream) {
  (void)in_sizes; (void)n_in; (void)out_size; (void)ws_size;
  const float* dec    = (const float*)d_in[0];
  const float* enc    = (const float*)d_in[1];
  const float* sa_Wq  = (const float*)d_in[2];
  const float* sa_Wk  = (const float*)d_in[3];
  const float* sa_Wv  = (const float*)d_in[4];
  const float* sa_Wo  = (const float*)d_in[5];
  const float* sa_g   = (const float*)d_in[6];
  const float* sa_b   = (const float*)d_in[7];
  const float* sa_rot = (const float*)d_in[8];
  const float* ca_Wq  = (const float*)d_in[9];
  const float* ca_Wk  = (const float*)d_in[10];
  const float* ca_Wv  = (const float*)d_in[11];
  const float* ca_Wo  = (const float*)d_in[12];
  const float* ca_g   = (const float*)d_in[13];
  const float* ca_b   = (const float*)d_in[14];
  const float* ca_rot = (const float*)d_in[15];
  const float* ff_W1  = (const float*)d_in[16];
  const float* ff_b1  = (const float*)d_in[17];
  const float* ff_W2  = (const float*)d_in[18];
  const float* ff_b2  = (const float*)d_in[19];
  const float* ff_g   = (const float*)d_in[20];
  const float* ff_b   = (const float*)d_in[21];

  char* base = (char*)d_ws;
  const size_t MB = 1ull << 20;
  Ws w;
  w.dec16 = (_Float16*)(base + 0 * MB);     // 16MB
  w.enc16 = (_Float16*)(base + 16 * MB);    // 16MB
  _Float16* saWq16 = (_Float16*)(base + 32 * MB);          // 0.5MB each
  _Float16* saWk16 = (_Float16*)(base + 32 * MB + 512 * 1024);
  _Float16* saWv16 = (_Float16*)(base + 33 * MB);
  _Float16* saWo16 = (_Float16*)(base + 33 * MB + 512 * 1024);
  _Float16* caWq16 = (_Float16*)(base + 34 * MB);
  _Float16* caWk16 = (_Float16*)(base + 34 * MB + 512 * 1024);
  _Float16* caWv16 = (_Float16*)(base + 35 * MB);
  _Float16* caWo16 = (_Float16*)(base + 35 * MB + 512 * 1024);
  _Float16* ffW116 = (_Float16*)(base + 36 * MB);          // 2MB
  _Float16* ffW216 = (_Float16*)(base + 38 * MB);          // 2MB
  w.Q16  = (_Float16*)(base + 40 * MB);     // 16MB \
  w.K16  = (_Float16*)(base + 56 * MB);     // 16MB  } Hb16 (64MB) aliases these
  w.V16  = (_Float16*)(base + 72 * MB);     // 16MB  } during the FFN
  w.O16  = (_Float16*)(base + 88 * MB);     // 16MB /
  w.Hb16 = (_Float16*)(base + 40 * MB);
  w.X1    = (float*)(base + 104 * MB);      // 32MB
  w.X1_16 = (_Float16*)(base + 136 * MB);   // 16MB
  w.X2    = (float*)(base + 152 * MB);      // 32MB
  w.X2_16 = (_Float16*)(base + 168 * MB);   // 16MB
  w.TMP   = (float*)(base + 184 * MB);      // 32MB
  w.bq = (int*)(base + 216 * MB);
  w.bk = (int*)(base + 217 * MB);
  w.pq = (int*)(base + 218 * MB);
  w.pk = (int*)(base + 219 * MB);

  auto cvt = [&](const float* s, _Float16* d, int n) {
    cvt_kernel<<<(n / 4 + 255) / 256, 256, 0, stream>>>(s, d, n / 4);
  };
  cvt(dec, w.dec16, RBS * RD);
  cvt(enc, w.enc16, RBS * RD);
  cvt(sa_Wq, saWq16, RD * RD);  cvt(sa_Wk, saWk16, RD * RD);
  cvt(sa_Wv, saWv16, RD * RD);  cvt(sa_Wo, saWo16, RD * RD);
  cvt(ca_Wq, caWq16, RD * RD);  cvt(ca_Wk, caWk16, RD * RD);
  cvt(ca_Wv, caWv16, RD * RD);  cvt(ca_Wo, caWo16, RD * RD);
  cvt(ff_W1, ffW116, RD * 4 * RD);
  cvt(ff_W2, ffW216, 4 * RD * RD);

  // Self-attention (LSH) -> cross-attention (LSH) -> FFN.
  run_lsh_attention(w.dec16, w.dec16, dec, saWq16, saWk16, saWv16, saWo16,
                    sa_g, sa_b, sa_rot, w.X1, w.X1_16, w, stream);
  run_lsh_attention(w.X1_16, w.enc16, w.X1, caWq16, caWk16, caWv16, caWo16,
                    ca_g, ca_b, ca_rot, w.X2, w.X2_16, w, stream);

  // FFN: Hb = relu(X2*W1 + b1) (f16); TMP = X2 + Hb*W2 + b2; out = LN(TMP)
  launch_gemm(w.X2_16, ffW116, ff_b1, nullptr, nullptr, w.Hb16,
              RBS, 4 * RD, RD, 1, stream);
  launch_gemm(w.Hb16, ffW216, ff_b2, /*resid=*/w.X2, w.TMP, nullptr,
              RBS, RD, 4 * RD, 0, stream);
  ln_kernel<<<RBS, 256, 0, stream>>>(w.TMP, ff_g, ff_b, (float*)d_out, nullptr);
}
